// EEGConfusionNet_85830626443966
// MI455X (gfx1250) — matrix operand order repaired and checked by reading, not compiled, not run on hardware
//
#include <hip/hip_runtime.h>
#include <hip/hip_bf16.h>
#include <math.h>

// ---------------- model dims ----------------
#define B_   32
#define CIN  8
#define T_   500
#define D_   192      // 3*64
#define NH_  4
#define HD_  48
#define H_   128
#define MT   (B_*T_)  // 16000 rows
#define EPSF 1e-5f

typedef __attribute__((ext_vector_type(16))) _Float16 v16h;
typedef __attribute__((ext_vector_type(8)))  _Float16 v8h;
typedef __attribute__((ext_vector_type(4)))  _Float16 v4h;
typedef __attribute__((ext_vector_type(8)))  float    v8f;

__device__ __forceinline__ float sigmoidf_(float x) { return 1.0f / (1.0f + __expf(-x)); }

__device__ __forceinline__ v4h cvt4(float4 v) {
  v4h h;
  h[0] = (_Float16)v.x; h[1] = (_Float16)v.y;
  h[2] = (_Float16)v.z; h[3] = (_Float16)v.w;
  return h;
}

// =====================================================================
// 1) Multi-scale conv (k=3,5,7) + ReLU, output transposed to A[m,f]
// =====================================================================
__global__ void k_conv(const float* __restrict__ x,
                       const float* __restrict__ w3, const float* __restrict__ b3,
                       const float* __restrict__ w5, const float* __restrict__ b5,
                       const float* __restrict__ w7, const float* __restrict__ b7,
                       float* __restrict__ A) {
  int m = blockIdx.x;          // b*T + t
  int f = threadIdx.x;         // 0..191
  int b = m / T_, t = m % T_;
  int g = f / 64, fl = f % 64;
  const float* w; const float* bb; int k;
  if (g == 0)      { w = w3; bb = b3; k = 3; }
  else if (g == 1) { w = w5; bb = b5; k = 5; }
  else             { w = w7; bb = b7; k = 7; }
  const float* wf = w + fl * CIN * k;
  int half = k / 2;
  float acc = bb[fl];
  for (int c = 0; c < CIN; ++c) {
    const float* xr = x + (size_t)(b * CIN + c) * T_;
    for (int dk = 0; dk < k; ++dk) {
      int tt = t + dk - half;
      if (tt >= 0 && tt < T_) acc += xr[tt] * wf[c * k + dk];
    }
  }
  A[(size_t)m * D_ + f] = fmaxf(acc, 0.0f);
}

// =====================================================================
// 2) BatchNorm stats per channel (biased var) -> fused scale/shift
// =====================================================================
__global__ void k_bnstats(const float* __restrict__ A, const float* __restrict__ gw,
                          const float* __restrict__ gb, float* __restrict__ scale,
                          float* __restrict__ shift) {
  int f = blockIdx.x; int tid = threadIdx.x;
  __shared__ float s1[256], s2[256];
  float a = 0.f, b = 0.f;
  for (int m = tid; m < MT; m += 256) {
    float v = A[(size_t)m * D_ + f];
    a += v; b += v * v;
  }
  s1[tid] = a; s2[tid] = b; __syncthreads();
  for (int o = 128; o > 0; o >>= 1) {
    if (tid < o) { s1[tid] += s1[tid + o]; s2[tid] += s2[tid + o]; }
    __syncthreads();
  }
  if (tid == 0) {
    float mu  = s1[0] / (float)MT;
    float var = s2[0] / (float)MT - mu * mu;
    float sc  = gw[f] * rsqrtf(var + EPSF);
    scale[f] = sc;
    shift[f] = gb[f] - mu * sc;
  }
}

__global__ void k_bnapply(float* __restrict__ A, const float* __restrict__ scale,
                          const float* __restrict__ shift) {
  size_t i = (size_t)blockIdx.x * 256 + threadIdx.x;
  if (i >= (size_t)MT * D_) return;
  int f = (int)(i % D_);
  A[i] = A[i] * scale[f] + shift[f];
}

// =====================================================================
// 3) WMMA GEMM:  C[M,N] = A[M,K] * W[N,K]^T + bias0 + bias1
//    Requirements (all call sites satisfy): M%32==0, N%64==0, K%32==0.
//    Block = 128 thr (4 waves); block tile 32(M) x 64(N); each wave owns a
//    16-col slab and two 16x16 sub-tiles (2 wmma per K step, B frag reuse).
//    fp32 -> fp16 staged via LDS with float4 loads / 8B LDS stores; fragment
//    gathers are contiguous 16B ds_load_b128s (layouts keep K contiguous).
// =====================================================================
__global__ void k_gemm_wmma(const float* __restrict__ A, const float* __restrict__ W,
                            const float* __restrict__ bias0, const float* __restrict__ bias1,
                            float* __restrict__ C, int M, int N, int K) {
  __shared__ __attribute__((aligned(16))) _Float16 As[32][40];      // rows of A, K-contig
  __shared__ __attribute__((aligned(16))) _Float16 Ws[4][16][40];   // rows of W, K-contig
  int wave = threadIdx.x >> 5;
  int lane = threadIdx.x & 31;
  int m0 = blockIdx.x * 32;
  int n0 = blockIdx.y * 64 + wave * 16;

  int r  = lane & 15;
  int kb = (lane < 16) ? 0 : 8;            // A fragment K base (ISA layout)
  int kk = (lane < 16) ? 0 : 16;           // unused; B fragment uses kb
  int nn = lane & 15;

  v8f acc0 = {}, acc1 = {};
  for (int k0 = 0; k0 < K; k0 += 32) {
    if (k0 + 32 < K) {   // prefetch next K tile (global_prefetch_b8)
      __builtin_prefetch(A + (size_t)(m0 + (threadIdx.x & 31)) * K + k0 + 32, 0, 3);
      __builtin_prefetch(W + (size_t)(n0 + (lane & 15)) * K + k0 + 32, 0, 3);
    }
    // stage A tile 32x32 (256 float4 over 128 threads)
    #pragma unroll
    for (int q = 0; q < 2; ++q) {
      int idx = threadIdx.x + q * 128;     // 0..255
      int rr = idx >> 3, c4 = idx & 7;
      float4 v = *(const float4*)(A + (size_t)(m0 + rr) * K + k0 + c4 * 4);
      *(v4h*)&As[rr][c4 * 4] = cvt4(v);
    }
    // stage this wave's 16x32 W slab (128 float4 over 32 lanes)
    #pragma unroll
    for (int q = 0; q < 4; ++q) {
      int idx = lane + q * 32;             // 0..127
      int n = idx >> 3, c4 = idx & 7;
      float4 v = *(const float4*)(W + (size_t)(n0 + n) * K + k0 + c4 * 4);
      *(v4h*)&Ws[wave][n][c4 * 4] = cvt4(v);
    }
    __syncthreads();
    // B fragment: column nn, K = kb..kb+7 and kb+16..kb+23 (ISA layout)
    v8h blo = *(const v8h*)&Ws[wave][nn][kb];
    v8h bhi = *(const v8h*)&Ws[wave][nn][kb + 16];
    v16h bf = __builtin_shufflevector(blo, bhi, 0,1,2,3,4,5,6,7,8,9,10,11,12,13,14,15);
    // A fragments for sub-tiles m0..m0+15 and m0+16..m0+31
    v8h a0lo = *(const v8h*)&As[r][kb];
    v8h a0hi = *(const v8h*)&As[r][kb + 16];
    v16h a0 = __builtin_shufflevector(a0lo, a0hi, 0,1,2,3,4,5,6,7,8,9,10,11,12,13,14,15);
    v8h a1lo = *(const v8h*)&As[16 + r][kb];
    v8h a1hi = *(const v8h*)&As[16 + r][kb + 16];
    v16h a1 = __builtin_shufflevector(a1lo, a1hi, 0,1,2,3,4,5,6,7,8,9,10,11,12,13,14,15);
    acc0 = __builtin_amdgcn_wmma_f32_16x16x32_f16(false, a0, false, bf, (short)0, acc0, false, false);
    acc1 = __builtin_amdgcn_wmma_f32_16x16x32_f16(false, a1, false, bf, (short)0, acc1, false, false);
    __syncthreads();
  }
  int rb = (lane < 16) ? 0 : 8;
  int ng = n0 + nn;
  float bs = (bias0 ? bias0[ng] : 0.f) + (bias1 ? bias1[ng] : 0.f);
  #pragma unroll
  for (int v = 0; v < 8; ++v) {
    C[(size_t)(m0 + rb + v) * N + ng]      = acc0[v] + bs;
    C[(size_t)(m0 + 16 + rb + v) * N + ng] = acc1[v] + bs;
  }
}

// =====================================================================
// 4) GLU combine: h = val * sigmoid(gate)
// =====================================================================
__global__ void k_glu(const float* __restrict__ g, float* __restrict__ h) {
  size_t i = (size_t)blockIdx.x * 256 + threadIdx.x;
  if (i >= (size_t)MT * D_) return;
  size_t m = i / D_; int n = (int)(i % D_);
  float v  = g[m * (2 * D_) + n];
  float gt = g[m * (2 * D_) + D_ + n];
  h[i] = v * sigmoidf_(gt);
}

// =====================================================================
// 5) GAT per-node logits: a_s/a_d = <hh_head, a_src/dst_head>
// =====================================================================
__global__ void k_asd(const float* __restrict__ hh, const float* __restrict__ asrc,
                      const float* __restrict__ adst, float* __restrict__ a_s,
                      float* __restrict__ a_d) {
  int idx = blockIdx.x * 256 + threadIdx.x;  // m*NH + h
  if (idx >= MT * NH_) return;
  int m = idx / NH_, h = idx % NH_;
  const float* row = hh + (size_t)m * D_ + h * HD_;
  float s = 0.f, d = 0.f;
  for (int i = 0; i < HD_; ++i) {
    s += row[i] * asrc[h * HD_ + i];
    d += row[i] * adst[h * HD_ + i];
  }
  a_s[idx] = s; a_d[idx] = d;
}

// =====================================================================
// 6) GAT softmax row stats over destinations j (max + sumexp)
// =====================================================================
__global__ void k_gatstats(const float* __restrict__ a_s, const float* __restrict__ a_d,
                           float* __restrict__ rmax, float* __restrict__ rsum) {
  int idx = blockIdx.x * 256 + threadIdx.x;      // b*(NH*T) + h*T + i
  if (idx >= B_ * NH_ * T_) return;
  int b = idx / (NH_ * T_);
  int rem = idx % (NH_ * T_);
  int h = rem / T_, i = rem % T_;
  float s = a_s[(size_t)(b * T_ + i) * NH_ + h];
  float mx = -1e30f;
  for (int j = 0; j < T_; ++j) {
    float v = s + a_d[(size_t)(b * T_ + j) * NH_ + h];
    v = (v > 0.f) ? v : 0.2f * v;
    mx = fmaxf(mx, v);
  }
  float sm = 0.f;
  for (int j = 0; j < T_; ++j) {
    float v = s + a_d[(size_t)(b * T_ + j) * NH_ + h];
    v = (v > 0.f) ? v : 0.2f * v;
    sm += __expf(v - mx);
  }
  rmax[idx] = mx; rsum[idx] = sm;
}

// =====================================================================
// 7) GAT aggregation: out = softmax(P) @ hh  via WMMA (flash-style).
//    One wave per (b, head, 16-row tile); P tile built in registers.
// =====================================================================
__global__ void k_gatapply(const float* __restrict__ hh, const float* __restrict__ a_s,
                           const float* __restrict__ a_d, const float* __restrict__ rmax,
                           const float* __restrict__ rsum, float* __restrict__ out) {
  int i0 = blockIdx.x * 16;
  int hd = blockIdx.y;
  int b  = blockIdx.z;
  int lane = threadIdx.x;
  __shared__ float dsh[32];
  __shared__ __attribute__((aligned(16))) _Float16 hhs[32][HD_];   // 96B row stride

  int r  = lane & 15;
  int ig = i0 + r;
  bool iv = (ig < T_);
  float s = 0.f, rmx = 0.f, rinv = 0.f;
  if (iv) {
    s = a_s[(size_t)(b * T_ + ig) * NH_ + hd];
    int si = (b * NH_ + hd) * T_ + ig;
    rmx = rmax[si];
    rinv = 1.0f / rsum[si];
  }
  int kb = (lane < 16) ? 0 : 8;
  int kk = (lane < 16) ? 0 : 16;
  int nn = lane & 15;

  v8f acc0 = {}, acc1 = {}, acc2 = {};
  for (int jb = 0; jb < T_; jb += 32) {
    int jg = jb + lane;
    if (jg < T_) {
      dsh[lane] = a_d[(size_t)(b * T_ + jg) * NH_ + hd];
      const float4* src4 = (const float4*)(hh + (size_t)(b * T_ + jg) * D_ + hd * HD_);
      #pragma unroll
      for (int q = 0; q < HD_ / 4; ++q)
        *(v4h*)&hhs[lane][q * 4] = cvt4(src4[q]);
    } else {
      dsh[lane] = 0.f;
      v4h z = {};
      #pragma unroll
      for (int q = 0; q < HD_ / 4; ++q) *(v4h*)&hhs[lane][q * 4] = z;
    }
    __syncthreads();
    v16h p;
    #pragma unroll
    for (int ii = 0; ii < 8; ++ii) {
      int k1 = kb + ii, k2 = kb + 16 + ii;
      float p1 = 0.f, p2 = 0.f;
      if (iv && (jb + k1) < T_) {
        float v = s + dsh[k1]; v = (v > 0.f) ? v : 0.2f * v;
        p1 = __expf(v - rmx) * rinv;
      }
      if (iv && (jb + k2) < T_) {
        float v = s + dsh[k2]; v = (v > 0.f) ? v : 0.2f * v;
        p2 = __expf(v - rmx) * rinv;
      }
      p[ii] = (_Float16)p1; p[8 + ii] = (_Float16)p2;
    }
    v16h b0, b1, b2;
    #pragma unroll
    for (int ii = 0; ii < 16; ++ii) {
      b0[ii] = hhs[kb + (ii & 7) + ((ii >> 3) << 4)][nn];
      b1[ii] = hhs[kb + (ii & 7) + ((ii >> 3) << 4)][16 + nn];
      b2[ii] = hhs[kb + (ii & 7) + ((ii >> 3) << 4)][32 + nn];
    }
    acc0 = __builtin_amdgcn_wmma_f32_16x16x32_f16(false, p, false, b0, (short)0, acc0, false, false);
    acc1 = __builtin_amdgcn_wmma_f32_16x16x32_f16(false, p, false, b1, (short)0, acc1, false, false);
    acc2 = __builtin_amdgcn_wmma_f32_16x16x32_f16(false, p, false, b2, (short)0, acc2, false, false);
    __syncthreads();
  }
  int rb = (lane < 16) ? 0 : 8;
  #pragma unroll
  for (int v = 0; v < 8; ++v) {
    int mg = i0 + rb + v;
    if (mg < T_) {
      float* o = out + (size_t)(b * T_ + mg) * D_ + hd * HD_;
      o[nn] = acc0[v]; o[16 + nn] = acc1[v]; o[32 + nn] = acc2[v];
    }
  }
}

// =====================================================================
// 8) Channel attention: pooled mean, tiny MLP, channel scale
// =====================================================================
__global__ void k_pool(const float* __restrict__ h2, float* __restrict__ pooled) {
  int idx = blockIdx.x * 256 + threadIdx.x;
  if (idx >= B_ * D_) return;
  int b = idx / D_, d = idx % D_;
  float s = 0.f;
  for (int t = 0; t < T_; ++t) s += h2[(size_t)(b * T_ + t) * D_ + d];
  pooled[idx] = s / (float)T_;
}

__global__ void k_camlp(const float* __restrict__ pooled,
                        const float* __restrict__ w1, const float* __restrict__ b1,
                        const float* __restrict__ w2, const float* __restrict__ b2,
                        float* __restrict__ wca) {
  int b = blockIdx.x; int tid = threadIdx.x; // 192
  __shared__ float ps[192], z1[96];
  ps[tid] = pooled[b * D_ + tid];
  __syncthreads();
  if (tid < 96) {
    float a = b1[tid];
    for (int k = 0; k < D_; ++k) a += w1[tid * D_ + k] * ps[k];
    z1[tid] = fmaxf(a, 0.f);
  }
  __syncthreads();
  float a = b2[tid];
  for (int k = 0; k < 96; ++k) a += w2[tid * 96 + k] * z1[k];
  wca[b * D_ + tid] = sigmoidf_(a);
}

__global__ void k_scale(const float* __restrict__ h2, const float* __restrict__ wca,
                        float* __restrict__ h3) {
  size_t i = (size_t)blockIdx.x * 256 + threadIdx.x;
  if (i >= (size_t)MT * D_) return;
  int f = (int)(i % D_);
  int m = (int)(i / D_);
  int b = m / T_;
  h3[i] = h2[i] * wca[b * D_ + f];
}

// =====================================================================
// 9) LSTM recurrence (one block per batch element, 512 threads = 4H)
// =====================================================================
__global__ void k_lstm(const float* __restrict__ pre, const float* __restrict__ whh,
                       float* __restrict__ y, int reverse) {
  int b = blockIdx.x; int n = threadIdx.x; // 0..511
  __shared__ float hs[H_], cs[H_], gs[4 * H_];
  if (n < H_) { hs[n] = 0.f; cs[n] = 0.f; }
  __syncthreads();
  for (int st = 0; st < T_; ++st) {
    int t = reverse ? (T_ - 1 - st) : st;
    float a = pre[(size_t)(b * T_ + t) * (4 * H_) + n];
    const float* wr = whh + (size_t)n * H_;
    for (int k = 0; k < H_; ++k) a += wr[k] * hs[k];
    gs[n] = a;
    __syncthreads();
    if (n < H_) {
      float ig = sigmoidf_(gs[n]);
      float fg = sigmoidf_(gs[H_ + n]);
      float gg = tanhf(gs[2 * H_ + n]);
      float og = sigmoidf_(gs[3 * H_ + n]);
      float c  = fg * cs[n] + ig * gg;
      cs[n] = c;
      float h = og * tanhf(c);
      hs[n] = h;
      y[(size_t)(b * T_ + t) * H_ + n] = h;
    }
    __syncthreads();
  }
}

// =====================================================================
// 10) Temporal attention
// =====================================================================
__global__ void k_tascore(const float* __restrict__ hf, const float* __restrict__ hb,
                          const float* __restrict__ w1, const float* __restrict__ b1,
                          const float* __restrict__ w2, const float* __restrict__ b2,
                          float* __restrict__ score) {
  int m = blockIdx.x; int n = threadIdx.x; // 128
  __shared__ float red[128];
  const float* hfm = hf + (size_t)m * H_;
  const float* hbm = hb + (size_t)m * H_;
  float a = b1[n];
  for (int k = 0; k < H_; ++k) a += w1[n * (2 * H_) + k] * hfm[k];
  for (int k = 0; k < H_; ++k) a += w1[n * (2 * H_) + H_ + k] * hbm[k];
  red[n] = tanhf(a) * w2[n];
  __syncthreads();
  for (int o = 64; o > 0; o >>= 1) {
    if (n < o) red[n] += red[n + o];
    __syncthreads();
  }
  if (n == 0) score[m] = red[0] + b2[0];
}

__global__ void k_tactx(const float* __restrict__ score, const float* __restrict__ hf,
                        const float* __restrict__ hb, float* __restrict__ ctx) {
  int b = blockIdx.x; int tid = threadIdx.x; // 256
  __shared__ float red[256];
  __shared__ float smax, ssum;
  float mx = -1e30f;
  for (int t = tid; t < T_; t += 256) mx = fmaxf(mx, score[b * T_ + t]);
  red[tid] = mx; __syncthreads();
  for (int o = 128; o > 0; o >>= 1) {
    if (tid < o) red[tid] = fmaxf(red[tid], red[tid + o]);
    __syncthreads();
  }
  if (tid == 0) smax = red[0];
  __syncthreads();
  float sm = 0.f;
  for (int t = tid; t < T_; t += 256) sm += __expf(score[b * T_ + t] - smax);
  red[tid] = sm; __syncthreads();
  for (int o = 128; o > 0; o >>= 1) {
    if (tid < o) red[tid] += red[tid + o];
    __syncthreads();
  }
  if (tid == 0) ssum = red[0];
  __syncthreads();
  float inv = 1.0f / ssum;
  float acc = 0.f;
  for (int t = 0; t < T_; ++t) {
    float w = __expf(score[b * T_ + t] - smax) * inv;
    float hv = (tid < H_) ? hf[(size_t)(b * T_ + t) * H_ + tid]
                          : hb[(size_t)(b * T_ + t) * H_ + (tid - H_)];
    acc += w * hv;
  }
  ctx[b * (2 * H_) + tid] = acc;
}

// =====================================================================
// 11) GRN + LayerNorm + classifier (one block per batch element)
// =====================================================================
__global__ void k_final(const float* __restrict__ ctx,
                        const float* __restrict__ fc1w, const float* __restrict__ fc1b,
                        const float* __restrict__ fc2w, const float* __restrict__ fc2b,
                        const float* __restrict__ gw,   const float* __restrict__ gb,
                        const float* __restrict__ lng,  const float* __restrict__ lnb,
                        const float* __restrict__ c1w,  const float* __restrict__ c1b,
                        const float* __restrict__ c2w,  const float* __restrict__ c2b,
                        float* __restrict__ out) {
  int b = blockIdx.x; int tid = threadIdx.x; // 256
  __shared__ float cx[256], z1[128], zn[256], red[256], c1[128];
  cx[tid] = ctx[b * 256 + tid];
  __syncthreads();
  float v = 0.f;
  if (tid < 128) {
    float a = fc1b[tid];
    for (int k = 0; k < 256; ++k) a += fc1w[tid * 256 + k] * cx[k];
    v = (a > 0.f) ? a : (__expf(a) - 1.0f);   // ELU
  }
  __syncthreads();
  if (tid < 128) z1[tid] = v;
  __syncthreads();
  float a2 = fc2b[tid];
  for (int k = 0; k < 128; ++k) a2 += fc2w[tid * 128 + k] * z1[k];
  float g = gb[tid];
  for (int k = 0; k < 256; ++k) g += gw[tid * 256 + k] * cx[k];
  g = sigmoidf_(g);
  float z = g * a2 + (1.0f - g) * cx[tid];
  // LayerNorm over 256
  red[tid] = z; __syncthreads();
  for (int o = 128; o > 0; o >>= 1) { if (tid < o) red[tid] += red[tid + o]; __syncthreads(); }
  float mu = red[0] / 256.0f;
  __syncthreads();
  red[tid] = (z - mu) * (z - mu); __syncthreads();
  for (int o = 128; o > 0; o >>= 1) { if (tid < o) red[tid] += red[tid + o]; __syncthreads(); }
  float var = red[0] / 256.0f;
  zn[tid] = (z - mu) * rsqrtf(var + EPSF) * lng[tid] + lnb[tid];
  __syncthreads();
  if (tid < 128) {
    float a = c1b[tid];
    for (int k = 0; k < 256; ++k) a += c1w[tid * 256 + k] * zn[k];
    c1[tid] = fmaxf(a, 0.f);
  }
  __syncthreads();
  if (tid < 3) {
    float a = c2b[tid];
    for (int k = 0; k < 128; ++k) a += c2w[tid * 128 + k] * c1[k];
    out[b * 3 + tid] = a;
  }
}

// =====================================================================
// Host-side orchestration
// =====================================================================
extern "C" void kernel_launch(void* const* d_in, const int* in_sizes, int n_in,
                              void* d_out, int out_size, void* d_ws, size_t ws_size,
                              hipStream_t stream) {
  (void)in_sizes; (void)n_in; (void)out_size; (void)ws_size;
  const float* x       = (const float*)d_in[0];
  const float* conv3_w = (const float*)d_in[1];
  const float* conv3_b = (const float*)d_in[2];
  const float* conv5_w = (const float*)d_in[3];
  const float* conv5_b = (const float*)d_in[4];
  const float* conv7_w = (const float*)d_in[5];
  const float* conv7_b = (const float*)d_in[6];
  const float* bn_g    = (const float*)d_in[7];
  const float* bn_b    = (const float*)d_in[8];
  const float* glu_w   = (const float*)d_in[9];
  const float* glu_b   = (const float*)d_in[10];
  const float* gat_w   = (const float*)d_in[11];
  const float* a_src   = (const float*)d_in[12];
  const float* a_dst   = (const float*)d_in[13];
  const float* ca_w1   = (const float*)d_in[14];
  const float* ca_b1   = (const float*)d_in[15];
  const float* ca_w2   = (const float*)d_in[16];
  const float* ca_b2   = (const float*)d_in[17];
  const float* lf_wih  = (const float*)d_in[18];
  const float* lf_whh  = (const float*)d_in[19];
  const float* lf_bih  = (const float*)d_in[20];
  const float* lf_bhh  = (const float*)d_in[21];
  const float* lb_wih  = (const float*)d_in[22];
  const float* lb_whh  = (const float*)d_in[23];
  const float* lb_bih  = (const float*)d_in[24];
  const float* lb_bhh  = (const float*)d_in[25];
  const float* ta_w1   = (const float*)d_in[26];
  const float* ta_b1   = (const float*)d_in[27];
  const float* ta_w2   = (const float*)d_in[28];
  const float* ta_b2   = (const float*)d_in[29];
  const float* fc1w    = (const float*)d_in[30];
  const float* fc1b    = (const float*)d_in[31];
  const float* fc2w    = (const float*)d_in[32];
  const float* fc2b    = (const float*)d_in[33];
  const float* gatew   = (const float*)d_in[34];
  const float* gateb   = (const float*)d_in[35];
  const float* lngv    = (const float*)d_in[36];
  const float* lnbv    = (const float*)d_in[37];
  const float* c1w     = (const float*)d_in[38];
  const float* c1b     = (const float*)d_in[39];
  const float* c2w     = (const float*)d_in[40];
  const float* c2b     = (const float*)d_in[41];

  float* ws = (float*)d_ws;
  // workspace layout (floats); total ~32.04M floats ~= 128.2 MB
  float* A     = ws + 0;               // [MT,192] conv out -> BN'd; later h3
  float* gbuf  = ws + 3072000;         // [MT,384] GLU pre;   later hf/hb
  float* hbuf  = ws + 9216000;         // [MT,192] post-GLU;  later GAT out h2
  float* hhbuf = ws + 12288000;        // [MT,192] GAT projection
  float* asb   = ws + 15360000;        // [MT*4]
  float* adb   = ws + 15424000;        // [MT*4]
  float* rmaxb = ws + 15488000;        // [B*NH*T]
  float* rsumb = ws + 15552000;        // [B*NH*T]
  float* scl   = ws + 15616000;        // [192]
  float* shf   = ws + 15616192;        // [192]
  float* pool  = ws + 15616384;        // [B*192]
  float* wca   = ws + 15622528;        // [B*192]
  float* score = ws + 15628672;        // [MT]
  float* ctx   = ws + 15644672;        // [B*256]
  float* pre_f = ws + 15652864;        // [MT,512]
  float* pre_b = ws + 23844864;        // [MT,512]
  float* h2    = hbuf;                 // alias (h dead after projection+logits)
  float* h3    = A;                    // alias (A dead after GLU GEMM)
  float* hf    = gbuf;                 // alias (g dead after GLU combine)
  float* hb    = gbuf + 2048000;

  const size_t nMD = (size_t)MT * D_;

  // 1) conv + relu (writes A[m,f])
  k_conv<<<MT, D_, 0, stream>>>(x, conv3_w, conv3_b, conv5_w, conv5_b,
                                conv7_w, conv7_b, A);
  // 2) BN stats + apply
  k_bnstats<<<D_, 256, 0, stream>>>(A, bn_g, bn_b, scl, shf);
  k_bnapply<<<(int)((nMD + 255) / 256), 256, 0, stream>>>(A, scl, shf);
  // 3) GLU GEMM (WMMA): g = A * glu_w^T + glu_b   [MT,384]
  k_gemm_wmma<<<dim3(MT / 32, 6), 128, 0, stream>>>(A, glu_w, glu_b, nullptr,
                                                    gbuf, MT, 2 * D_, D_);
  k_glu<<<(int)((nMD + 255) / 256), 256, 0, stream>>>(gbuf, hbuf);
  // 4) GAT projection (WMMA): hh = h * gat_w^T    [MT,192]
  k_gemm_wmma<<<dim3(MT / 32, 3), 128, 0, stream>>>(hbuf, gat_w, nullptr, nullptr,
                                                    hhbuf, MT, D_, D_);
  // 5) GAT logits + row stats + WMMA aggregation
  k_asd<<<(MT * NH_ + 255) / 256, 256, 0, stream>>>(hhbuf, a_src, a_dst, asb, adb);
  k_gatstats<<<(B_ * NH_ * T_ + 255) / 256, 256, 0, stream>>>(asb, adb, rmaxb, rsumb);
  k_gatapply<<<dim3((T_ + 15) / 16, NH_, B_), 32, 0, stream>>>(hhbuf, asb, adb,
                                                               rmaxb, rsumb, h2);
  // 6) channel attention
  k_pool<<<(B_ * D_ + 255) / 256, 256, 0, stream>>>(h2, pool);
  k_camlp<<<B_, D_, 0, stream>>>(pool, ca_w1, ca_b1, ca_w2, ca_b2, wca);
  k_scale<<<(int)((nMD + 255) / 256), 256, 0, stream>>>(h2, wca, h3);
  // 7) LSTM input projections (WMMA) + recurrences
  k_gemm_wmma<<<dim3(MT / 32, 8), 128, 0, stream>>>(h3, lf_wih, lf_bih, lf_bhh,
                                                    pre_f, MT, 4 * H_, D_);
  k_gemm_wmma<<<dim3(MT / 32, 8), 128, 0, stream>>>(h3, lb_wih, lb_bih, lb_bhh,
                                                    pre_b, MT, 4 * H_, D_);
  k_lstm<<<B_, 4 * H_, 0, stream>>>(pre_f, lf_whh, hf, 0);
  k_lstm<<<B_, 4 * H_, 0, stream>>>(pre_b, lb_whh, hb, 1);
  // 8) temporal attention
  k_tascore<<<MT, H_, 0, stream>>>(hf, hb, ta_w1, ta_b1, ta_w2, ta_b2, score);
  k_tactx<<<B_, 2 * H_, 0, stream>>>(score, hf, hb, ctx);
  // 9) GRN + LN + classifier
  k_final<<<B_, 2 * H_, 0, stream>>>(ctx, fc1w, fc1b, fc2w, fc2b, gatew, gateb,
                                     lngv, lnbv, c1w, c1b, c2w, c2b, (float*)d_out);
}
